// StochasticDifferentialEquation_23210003267930
// MI455X (gfx1250) — matrix operand order repaired and checked by reading, not compiled, not run on hardware
//
#include <hip/hip_runtime.h>
#include <hip/hip_bf16.h>

typedef __attribute__((ext_vector_type(16))) _Float16 v16h;
typedef __attribute__((ext_vector_type(8)))  _Float16 v8h;
typedef __attribute__((ext_vector_type(8)))  float    v8f;

#define B_ROWS   1024
#define DDIM     256
#define NSTEPS   1000
#define HSTRIDE  264   // halfs per LDS row: 256 + 8 pad -> 4-bank shift/row, conflict-free b128

__global__ __launch_bounds__(512) void sde_em_wmma_kernel(
    const float* __restrict__ x0,
    const float* __restrict__ A,
    const float* __restrict__ sigma,
    const float* __restrict__ dW,
    float* __restrict__ out)
{
    // Shared f16 image of this WG's 16-row x slab, laid out for direct
    // A-fragment b128 loads. The f32 master state lives in registers.
    __shared__ __align__(16) _Float16 xh[16 * HSTRIDE];

    const int rowBase = blockIdx.x * 16;          // 64 WGs * 16 rows = 1024
    const int wave = threadIdx.x >> 5;            // 16 waves, each owns 16 output cols
    const int lane = threadIdx.x & 31;
    const int n16  = lane & 15;
    const int hi   = lane >> 4;                   // lane half selects K sub-range
    const int col  = (wave << 4) + n16;           // B/C/D fragment column: N = lane%16
    const int m    = n16;                         // A fragment row: M = lane%16

    // ---------------------------------------------------------------
    // Resident B fragments: A^T (256x16 slab for this wave) in f16.
    // B layout (32x16 f16): lane holds col N=lane%16; K = 32*ch + hi*8 + {0..7} and hi*8 + {16..23}.
    // A^T[k][n] == A[n][k] -> row `col` of A, contiguous in k. Built once.
    // ---------------------------------------------------------------
    v16h bfrag[8];
    {
        const float* arow = A + (size_t)col * DDIM;
#pragma unroll
        for (int ch = 0; ch < 8; ++ch) {
            const int kb = ch * 32 + hi * 8;
            float t[16];
            *(float4*)(t + 0)  = *(const float4*)(arow + kb + 0);
            *(float4*)(t + 4)  = *(const float4*)(arow + kb + 4);
            *(float4*)(t + 8)  = *(const float4*)(arow + kb + 16);
            *(float4*)(t + 12) = *(const float4*)(arow + kb + 20);
            v16h b;
#pragma unroll
            for (int j = 0; j < 16; ++j) b[j] = (_Float16)t[j];
            bfrag[ch] = b;
        }
    }

    const float dt  = 0.001f;
    const float sig = sigma[col] * 0.0316227766016838f;   // sigma * sqrt(dt)

    // ---- register-resident f32 state: element r lives at (rowBase+r+8*hi, col) ----
    // (matches the WMMA C/D layout, so drift lands directly on its state element)
    float xreg[8];
#pragma unroll
    for (int r = 0; r < 8; ++r) {
        const int mr = r + hi * 8;
        const size_t g = (size_t)(rowBase + mr) * DDIM + col;
        const float v = x0[g];
        xreg[r] = v;
        out[g] = v;                               // path point 0
        xh[mr * HSTRIDE + col] = (_Float16)v;     // f16 image for fragment loads
    }
    __syncthreads();

    for (int s = 0; s < NSTEPS; ++s) {
        // Prefetch next step's contiguous 64KB dW span from this WG's rows: 512 threads x 128B.
        if (s + 1 < NSTEPS) {
            const float* nxt = dW + ((size_t)(s + 1) * B_ROWS + rowBase) * DDIM;
            __builtin_prefetch(nxt + (size_t)threadIdx.x * 32, 0, 3);
        }

        // ---- drift tile = X[16x256] @ (A^T slab)[256x16], f32 accum ----
        v8f c = {};
#pragma unroll
        for (int ch = 0; ch < 8; ++ch) {
            // A layout (16x32 f16): lane row M=lane%16;
            // K = 32*ch + kb + {0..7} and kb + {16..23}, kb = hi ? 8 : 0
            // -> two contiguous 16-byte runs of packed f16: zero conversions.
            const int kb = ch * 32 + hi * 8;
            const _Float16* xr = xh + m * HSTRIDE + kb;
            const v8h lo = *(const v8h*)(xr);
            const v8h hh = *(const v8h*)(xr + 16);
            const v16h a = __builtin_shufflevector(
                lo, hh, 0, 1, 2, 3, 4, 5, 6, 7, 8, 9, 10, 11, 12, 13, 14, 15);
            c = __builtin_amdgcn_wmma_f32_16x16x32_f16(
                    false, a, false, bfrag[ch], (short)0, c, false, false);
        }

        __syncthreads();   // all fragment reads of xh complete before updates

        // ---- Euler-Maruyama update + stream path point ----
        const size_t dwBase  = ((size_t)s * B_ROWS + rowBase) * DDIM;
        const size_t outBase = dwBase + (size_t)B_ROWS * DDIM;   // step s+1
#pragma unroll
        for (int r = 0; r < 8; ++r) {
            const int mr = r + hi * 8;
            const float dw   = dW[dwBase + (size_t)mr * DDIM + col];
            const float xnew = fmaf(dt, c[r], fmaf(sig, dw, xreg[r]));
            xreg[r] = xnew;
            xh[mr * HSTRIDE + col] = (_Float16)xnew;
            out[outBase + (size_t)mr * DDIM + col] = xnew;
        }
        __syncthreads();   // xh updates visible before next step's fragment reads
    }
}

extern "C" void kernel_launch(void* const* d_in, const int* in_sizes, int n_in,
                              void* d_out, int out_size, void* d_ws, size_t ws_size,
                              hipStream_t stream) {
    const float* x0    = (const float*)d_in[0];   // [1024, 256]
    const float* A     = (const float*)d_in[1];   // [256, 256]
    const float* sigma = (const float*)d_in[2];   // [256]
    const float* dW    = (const float*)d_in[3];   // [1000, 1024, 256]
    float* out = (float*)d_out;                   // [1001, 1024, 256]

    dim3 grid(B_ROWS / 16);
    dim3 block(512);
    hipLaunchKernelGGL(sde_em_wmma_kernel, grid, block, 0, stream,
                       x0, A, sigma, dW, out);
}
